// MixedPatchEmbed_20486994002011
// MI455X (gfx1250) — compile-verified, matrix operand order checked by reading, not run on hardware
//
#include <hip/hip_runtime.h>
#include <hip/hip_bf16.h>

// ---------------------------------------------------------------------------
// MixedPatchEmbed for MI455X (gfx1250): masked im2col GEMMs via v_wmma_f32_16x16x32_f16
// ---------------------------------------------------------------------------

typedef _Float16 v16h __attribute__((ext_vector_type(16)));
typedef _Float16 v8h  __attribute__((ext_vector_type(8)));
typedef _Float16 v4h  __attribute__((ext_vector_type(4)));
typedef float    v8f  __attribute__((ext_vector_type(8)));
typedef float    vf4  __attribute__((ext_vector_type(4)));

#define BATCH 128
#define DIM   768
#define IMG   224
#define THRESH 7.0f

// workspace layout (bytes)
#define WS_META    0            // 16 ints: [0]=total16 [1]=total32 [2]=Ntotal
#define WS_LIST16  64           // 25088 * int2
#define WS_LIST32  200768       // 6272  * int2
#define WS_W16H    250944       // 589824 _Float16
#define WS_W32H    1430592      // 2359296 _Float16

// ---------------------------------------------------------------------------
// Kernel 1: masks, scans, seqlens/cls-index tails, cls rows, compact work lists
// ---------------------------------------------------------------------------
__global__ __launch_bounds__(128)
void setup_kernel(const float* __restrict__ entropy,    // [128,7,7]
                  const float* __restrict__ cls_token,  // [768]
                  float* __restrict__ out,
                  int* __restrict__ meta,
                  int2* __restrict__ list16,
                  int2* __restrict__ list32)
{
    const int b = threadIdx.x;   // one thread per batch sample
    unsigned long long m32 = 0ull;
    for (int i = 0; i < 49; ++i)
        if (entropy[b * 49 + i] < THRESH) m32 |= (1ull << i);

    const int c32 = __popcll(m32);          // coarse tokens kept
    const int c16 = 4 * (49 - c32);         // fine tokens kept
    const int seq = 1 + c16 + c32;

    __shared__ int s_seq[128], s_c16[128], s_c32[128];
    s_seq[b] = seq; s_c16[b] = c16; s_c32[b] = c32;
    __syncthreads();

    int base = 0, o16 = 0, o32 = 0, tot = 0, t16 = 0, t32 = 0;
    for (int j = 0; j < 128; ++j) {
        if (j < b) { base += s_seq[j]; o16 += s_c16[j]; o32 += s_c32[j]; }
        tot += s_seq[j]; t16 += s_c16[j]; t32 += s_c32[j];
    }
    if (b == 0) { meta[0] = t16; meta[1] = t32; meta[2] = tot; }

    // tails: seqlens then cls_token_indices (stored as float per harness dtype)
    const size_t tail = (size_t)tot * DIM;
    out[tail + b] = (float)seq;
    out[tail + 128 + b] = (float)base;

    // cls token row (no positional embedding added in the reference)
    float* crow = out + (size_t)base * DIM;
    for (int j = 0; j < DIM; ++j) crow[j] = cls_token[j];

    // compact work lists: fine (16px) tokens first, then coarse (32px)
    int r16 = 0;
    for (int n = 0; n < 196; ++n) {
        const int gy = n / 14, gx = n % 14;
        const int cell = (gy >> 1) * 7 + (gx >> 1);
        if (!((m32 >> cell) & 1ull)) {
            list16[o16 + r16] = make_int2((b << 16) | n, base + 1 + r16);
            ++r16;
        }
    }
    int r32 = 0;
    for (int n = 0; n < 49; ++n) {
        if ((m32 >> n) & 1ull) {
            list32[o32 + r32] = make_int2((b << 16) | n, base + 1 + c16 + r32);
            ++r32;
        }
    }
}

// ---------------------------------------------------------------------------
// Kernel 2: fp32 -> f16 weight downcast (weights stay L2-resident afterwards)
// ---------------------------------------------------------------------------
__global__ void cvt_kernel(const float* __restrict__ src, _Float16* __restrict__ dst, int n)
{
    int i = blockIdx.x * blockDim.x + threadIdx.x;
    if (i < n) dst[i] = (_Float16)src[i];
}

// ---------------------------------------------------------------------------
// Kernels 3/4: masked im2col GEMM.
//   One block = one 16-row M-tile of the compacted token list.
//   Per K-chunk(32): cooperative gather x -> LDS (f16), ds_load_b128 A frags,
//   4 waves x 12 N-tiles -> 768 output dims, v_wmma_f32_16x16x32_f16.
// ---------------------------------------------------------------------------
template <int P>
__global__ __launch_bounds__(128)
void gemm_kernel(const float* __restrict__ x,      // [128,3,224,224]
                 const _Float16* __restrict__ wh,  // [768][K] f16
                 const float* __restrict__ bias,   // [768]
                 const float* __restrict__ pos,    // [Ntok][768]
                 const int2* __restrict__ list,
                 const int* __restrict__ meta, int meta_idx,
                 float* __restrict__ out)
{
    constexpr int K = 3 * P * P;       // 768 or 3072
    constexpr int G = IMG / P;         // 14 or 7
    const int total = meta[meta_idx];
    const int mtile = blockIdx.x;
    if (mtile * 16 >= total) return;

    const int tid  = threadIdx.x;
    const int lane = tid & 31;
    const int wave = tid >> 5;
    const int hi   = lane >> 4;        // 0: lanes 0-15, 1: lanes 16-31
    const int lcol = lane & 15;
    const int koff = hi * 8;           // K sub-group per ISA A/B layout

    __shared__ __align__(16) _Float16 sA[16 * 32];   // one 16x32 A chunk
    __shared__ int2 sList[16];

    if (tid < 16) {
        const int g = mtile * 16 + tid;
        sList[tid] = (g < total) ? list[g] : make_int2(-1, -1);
    }
    __syncthreads();

    // staging role: each thread gathers 4 consecutive k for one row
    const int srow = tid >> 3;            // 0..15
    const int sseg = (tid & 7) * 4;       // k-local: 0,4,...,28
    const int2 se  = sList[srow];
    const bool svalid = se.x >= 0;
    const int sb = se.x >> 16;
    const int sn = se.x & 0xFFFF;
    const int sgy = sn / G, sgx = sn % G;
    const float* sx = x + (((size_t)sb * 3) * IMG + (size_t)sgy * P) * IMG + (size_t)sgx * P;

    // A-fragment base: row m = lane%16 in LDS
    const _Float16* arow = &sA[lcol * 32];

    v8f acc[12];
#pragma unroll
    for (int i = 0; i < 12; ++i) acc[i] = (v8f){0, 0, 0, 0, 0, 0, 0, 0};

    for (int kc = 0; kc < K; kc += 32) {
        // ---- cooperative im2col gather + f32->f16 into LDS ----
        vf4 v = (vf4){0, 0, 0, 0};
        if (svalid) {
            const int k4  = kc + sseg;
            const int c   = k4 / (P * P);
            const int rem = k4 % (P * P);
            const int py  = rem / P;
            const int px  = rem % P;       // multiple of 4: stays within one image row
            v = *(const vf4*)(sx + ((size_t)c * IMG + py) * IMG + px);
        }
        v4h pack = { (_Float16)v.x, (_Float16)v.y, (_Float16)v.z, (_Float16)v.w };
        *(v4h*)(&sA[srow * 32 + sseg]) = pack;
        __syncthreads();

        // ---- A fragment per ISA layout: ds_load_b128 x2 ----
        const v8h a0 = *(const v8h*)(arow + koff);        // K = kc + koff .. +7
        const v8h a1 = *(const v8h*)(arow + 16 + koff);   // K = kc + 16 + koff .. +7
        const v16h a = __builtin_shufflevector(a0, a1, 0, 1, 2, 3, 4, 5, 6, 7,
                                                       8, 9, 10, 11, 12, 13, 14, 15);

        // ---- 12 N-tiles per wave ----
#pragma unroll
        for (int nt = 0; nt < 12; ++nt) {
            const int d = (wave * 12 + nt) * 16 + lcol;
            const _Float16* brow = wh + (size_t)d * K + kc;
            if (nt == 0 && kc + 32 < K)
                __builtin_prefetch(brow + 32, 0, 1);      // global_prefetch_b8 next chunk
            const v8h b0 = *(const v8h*)(brow + koff);
            const v8h b1 = *(const v8h*)(brow + 16 + koff);
            const v16h bf = __builtin_shufflevector(b0, b1, 0, 1, 2, 3, 4, 5, 6, 7,
                                                            8, 9, 10, 11, 12, 13, 14, 15);
            acc[nt] = __builtin_amdgcn_wmma_f32_16x16x32_f16(
                false, a, false, bf, (short)0, acc[nt], false, false);
        }
        __syncthreads();
    }

    // ---- epilogue: + bias + positional embedding, scatter to compacted rows ----
#pragma unroll
    for (int nt = 0; nt < 12; ++nt) {
        const int d  = (wave * 12 + nt) * 16 + lcol;
        const float bd = bias[d];
#pragma unroll
        for (int vv = 0; vv < 8; ++vv) {
            const int m = vv + hi * 8;                    // C/D layout: M = vgpr + 8*hi
            const int2 e = sList[m];
            if (e.x >= 0) {
                const int n = e.x & 0xFFFF;
                out[(size_t)e.y * DIM + d] = acc[nt][vv] + bd + pos[(size_t)n * DIM + d];
            }
        }
    }
}

// ---------------------------------------------------------------------------
extern "C" void kernel_launch(void* const* d_in, const int* in_sizes, int n_in,
                              void* d_out, int out_size, void* d_ws, size_t ws_size,
                              hipStream_t stream)
{
    const float* x     = (const float*)d_in[0];
    const float* ent   = (const float*)d_in[1];
    const float* w16   = (const float*)d_in[2];
    const float* b16   = (const float*)d_in[3];
    const float* w32   = (const float*)d_in[4];
    const float* b32   = (const float*)d_in[5];
    const float* pos16 = (const float*)d_in[6];
    const float* pos32 = (const float*)d_in[7];
    const float* cls   = (const float*)d_in[8];
    float* out = (float*)d_out;

    char* ws = (char*)d_ws;
    int*  meta = (int*)(ws + WS_META);
    int2* l16  = (int2*)(ws + WS_LIST16);
    int2* l32  = (int2*)(ws + WS_LIST32);
    _Float16* w16h = (_Float16*)(ws + WS_W16H);
    _Float16* w32h = (_Float16*)(ws + WS_W32H);

    setup_kernel<<<1, 128, 0, stream>>>(ent, cls, out, meta, l16, l32);

    cvt_kernel<<<(589824 + 255) / 256, 256, 0, stream>>>(w16, w16h, 589824);
    cvt_kernel<<<(2359296 + 255) / 256, 256, 0, stream>>>(w32, w32h, 2359296);

    // worst-case grids; blocks past the runtime total early-exit on meta[]
    gemm_kernel<16><<<dim3(25088 / 16), 128, 0, stream>>>(x, w16h, b16, pos16, l16, meta, 0, out);
    gemm_kernel<32><<<dim3(6272 / 16),  128, 0, stream>>>(x, w32h, b32, pos32, l32, meta, 1, out);
}
